// MultiheadAttention_32349693674252
// MI455X (gfx1250) — matrix operand order repaired and checked by reading, not compiled, not run on hardware
//
#include <hip/hip_runtime.h>
#include <hip/hip_bf16.h>

// ---------------------------------------------------------------------------
// MHA forward for MI455X (gfx1250, wave32, WMMA 16x16x32 f16, f32 accum).
//   H=16 heads, DK=DV=64, DM=1024, B=2, S=2048.
// f32 inputs -> f16 q/k/vT in workspace -> flash-attention in LDS
// -> f16 concat -> f32 output projection. Score matrix never hits HBM.
// f16 tile copies use GLOBAL_LOAD_ASYNC_TO_LDS_B128 (ASYNCcnt-tracked).
// GEMM staging: batched b128 loads (8 in flight), packed f16 LDS stores.
// ---------------------------------------------------------------------------

#define MHA_H  16
#define MHA_DK 64
#define MHA_DV 64
#define MHA_DM 1024
#define MHA_B  2
#define MHA_S  2048
#define MHA_BS (MHA_B * MHA_S)   // 4096

typedef __attribute__((ext_vector_type(16))) _Float16 v16h;
typedef __attribute__((ext_vector_type(8)))  _Float16 v8h;
typedef __attribute__((ext_vector_type(4)))  _Float16 v4h;
typedef __attribute__((ext_vector_type(2)))  _Float16 v2h;
typedef __attribute__((ext_vector_type(8)))  float    v8f;
typedef __attribute__((ext_vector_type(4)))  float    v4f;

// Two contiguous 16-byte runs -> one 16-element f16 fragment.
static __device__ __forceinline__ v16h load_frag(const _Float16* p0,
                                                 const _Float16* p1) {
  v8h a = *(const v8h*)p0;
  v8h b = *(const v8h*)p1;
  v16h r;
#pragma unroll
  for (int i = 0; i < 8; ++i) { r[i] = a[i]; r[i + 8] = b[i]; }
  return r;
}

static __device__ __forceinline__ v8f wmma_f16(v16h a, v16h b, v8f c) {
  return __builtin_amdgcn_wmma_f32_16x16x32_f16(false, a, false, b,
                                                (short)0, c, false, false);
}

// 16-byte global -> LDS copy through the async engine (no VGPR round trip).
// Generic LDS pointers carry the aperture in the high 32 bits; the low 32
// bits are the LDS byte offset the instruction wants.
static __device__ __forceinline__ void async_copy_b128(const _Float16* lds_dst,
                                                       const _Float16* gsrc) {
  unsigned lds_off = (unsigned)(unsigned long long)(size_t)lds_dst;
  unsigned long long gaddr = (unsigned long long)(size_t)gsrc;
  asm volatile("global_load_async_to_lds_b128 %0, %1, off"
               :: "v"(lds_off), "v"(gaddr)
               : "memory");
}

static __device__ __forceinline__ void async_wait0() {
  asm volatile("s_wait_asynccnt 0" ::: "memory");
}

// ---------------------------------------------------------------------------
// Kernel 1: per-head projection  out = (X @ W[h] + b[h]) * scale  (f16 out)
//   X: [BS, DM] f32.  W: [H, DM, 64] f32.  bias: [H, 64] f32.
//   transposed==0: out[((bb*H+h)*S + s)*64 + dk]   (q, k)
//   transposed==1: out[((bb*H+h)*64 + dk)*S + s]   (vT)
// Block = 128 threads (4 waves), 64x64 tile, K staged 64 wide per barrier.
// ---------------------------------------------------------------------------
__global__ __launch_bounds__(128) void mha_proj_kernel(
    const float* __restrict__ X, const float* __restrict__ W,
    const float* __restrict__ bias, _Float16* __restrict__ out,
    float scale, int transposed) {
  __shared__ _Float16 a_lds[64 * 72];   // row-major [row][k], stride 72
  __shared__ _Float16 bt_lds[64 * 72];  // N-major  [n][k],   stride 72

  const int tid  = threadIdx.x;
  const int wave = tid >> 5;
  const int lane = tid & 31;
  const int h    = blockIdx.y;
  const int row0 = blockIdx.x * 64;     // 64 rows per block; never crosses batch
  const float* Wh = W + (size_t)h * MHA_DM * 64;

  v8f acc[4] = {};

  const int arow  = wave * 16 + (lane & 15);
  const int kh8   = (lane >> 4) * 8;    // A frag K-group base
  const int nlane = lane & 15;
  const int kb16  = (lane >> 4) * 16;   // unused; B frags use kh8 too

  for (int kk = 0; kk < MHA_DM; kk += 64) {
    // --- A tile 64x64: batch 8 b128 loads, then convert + packed stores. ---
    v4f xa[8];
#pragma unroll
    for (int i = 0; i < 8; ++i) {
      int vi = i * 128 + tid;            // 1024 float4 chunks
      int r  = vi >> 4;                  // 16 chunks per 64-float row
      int c4 = (vi & 15) * 4;
      xa[i] = *(const v4f*)&X[(size_t)(row0 + r) * MHA_DM + kk + c4];
    }
#pragma unroll
    for (int i = 0; i < 8; ++i) {
      int vi = i * 128 + tid;
      int r  = vi >> 4;
      int c4 = (vi & 15) * 4;
      v4h hx;
#pragma unroll
      for (int j = 0; j < 4; ++j) hx[j] = (_Float16)xa[i][j];
      *(v4h*)&a_lds[r * 72 + c4] = hx;
    }
    // --- Bt tile: pairs of k-rows -> packed v2h transposed stores. ---
    v4f wa[8];
#pragma unroll
    for (int i = 0; i < 4; ++i) {
      int vi = i * 128 + tid;            // 512 k-pair chunks
      int k2 = (vi >> 4) * 2;
      int n4 = (vi & 15) * 4;
      wa[2 * i + 0] = *(const v4f*)&Wh[(size_t)(kk + k2) * 64 + n4];
      wa[2 * i + 1] = *(const v4f*)&Wh[(size_t)(kk + k2 + 1) * 64 + n4];
    }
#pragma unroll
    for (int i = 0; i < 4; ++i) {
      int vi = i * 128 + tid;
      int k2 = (vi >> 4) * 2;
      int n4 = (vi & 15) * 4;
#pragma unroll
      for (int j = 0; j < 4; ++j) {
        v2h p;
        p[0] = (_Float16)wa[2 * i + 0][j];
        p[1] = (_Float16)wa[2 * i + 1][j];
        *(v2h*)&bt_lds[(n4 + j) * 72 + k2] = p;
      }
    }
    if (kk + 64 < MHA_DM) {  // pull next chunk toward the caches
      __builtin_prefetch(&X[(size_t)(row0 + (tid >> 1)) * MHA_DM + kk + 64], 0, 1);
      __builtin_prefetch(&Wh[(size_t)(kk + 64 + (tid >> 1)) * 64], 0, 1);
    }
    __syncthreads();

#pragma unroll
    for (int ks = 0; ks < 2; ++ks) {
      const _Float16* ap = &a_lds[arow * 72 + ks * 32 + kh8];
      v16h af = load_frag(ap, ap + 16);
#pragma unroll
      for (int nt = 0; nt < 4; ++nt) {
        const _Float16* bp = &bt_lds[(nt * 16 + nlane) * 72 + ks * 32 + kh8];
        acc[nt] = wmma_f16(af, load_frag(bp, bp + 16), acc[nt]);
      }
    }
    __syncthreads();
  }

  // Epilogue: add bias, fold scale, store f16.
  const int rbase = (lane >> 4) * 8;
  const int bb    = row0 >> 11;         // row0 / S
#pragma unroll
  for (int nt = 0; nt < 4; ++nt) {
#pragma unroll
    for (int r = 0; r < 8; ++r) {
      int row = row0 + wave * 16 + rbase + r;
      int s   = row & (MHA_S - 1);
      int col = nt * 16 + nlane;
      float v = (acc[nt][r] + bias[h * 64 + col]) * scale;
      size_t idx = transposed
          ? (((size_t)bb * MHA_H + h) * 64 + col) * MHA_S + s
          : (((size_t)bb * MHA_H + h) * MHA_S + s) * 64 + col;
      out[idx] = (_Float16)v;
    }
  }
}

// ---------------------------------------------------------------------------
// Kernel 2: flash attention per (b,h).  Block = 128 threads / 64 query rows.
//   qh,kh: [B*H, S, 64] f16 (q pre-scaled).  vT: [B*H, 64, S] f16.
//   cat:   [B, S, H*64] f16.
// K/V tiles are staged with async-to-LDS (ASYNCcnt), then consumed by WMMA.
// ---------------------------------------------------------------------------
__global__ __launch_bounds__(128) void mha_attn_kernel(
    const _Float16* __restrict__ qh, const _Float16* __restrict__ kh,
    const _Float16* __restrict__ vT, _Float16* __restrict__ cat) {
  __shared__ _Float16 k_lds[64 * 72];      // [key][dk]
  __shared__ _Float16 v_lds[64 * 72];      // [dv][key]
  __shared__ _Float16 p_lds[4][16 * 72];   // per-wave P transpose staging

  const int tid  = threadIdx.x;
  const int wave = tid >> 5;
  const int lane = tid & 31;
  const int bh   = blockIdx.y;
  const int q0   = blockIdx.x * 64;

  const _Float16* qp = qh + (size_t)bh * MHA_S * 64;
  const _Float16* kp = kh + (size_t)bh * MHA_S * 64;
  const _Float16* vp = vT + (size_t)bh * 64 * MHA_S;

  const int nlane = lane & 15;
  const int kh8   = (lane >> 4) * 8;
  const int kb16  = (lane >> 4) * 16;

  // Load this wave's 16 query rows as two A fragments (K = 0..31, 32..63).
  const int arow = q0 + wave * 16 + (lane & 15);
  v16h aq[2];
#pragma unroll
  for (int ks = 0; ks < 2; ++ks) {
    const _Float16* p = qp + (size_t)arow * 64 + ks * 32 + kh8;
    aq[ks] = load_frag(p, p + 16);
  }

  float mrow[8], lrow[8];
  v8f oacc[4] = {};
#pragma unroll
  for (int r = 0; r < 8; ++r) { mrow[r] = -1e30f; lrow[r] = 0.f; }

  for (int kb = 0; kb < MHA_S; kb += 64) {
    // Async-stage 64 keys of K ([key][dk]) and V ([dv][key]) into LDS.
#pragma unroll
    for (int i = 0; i < 4; ++i) {
      int c = i * 128 + tid;             // 512 x 16B chunks per tile
      int r = c >> 3;                    // 8 chunks per 64-f16 row
      int o = (c & 7) * 8;
      async_copy_b128(&k_lds[r * 72 + o], &kp[(size_t)(kb + r) * 64 + o]);
      async_copy_b128(&v_lds[r * 72 + o], &vp[(size_t)r * MHA_S + kb + o]);
    }
    async_wait0();       // each wave drains its own ASYNCcnt ...
    __syncthreads();     // ... then the block barrier covers all waves

    // Scores: 16 query rows x 64 keys = 4 C-frags, 2 K-steps each.
    v8f sc[4] = {};
#pragma unroll
    for (int ks = 0; ks < 2; ++ks) {
#pragma unroll
      for (int nt = 0; nt < 4; ++nt) {
        const _Float16* bp = &k_lds[(nt * 16 + nlane) * 72 + ks * 32 + kh8];
        sc[nt] = wmma_f16(aq[ks], load_frag(bp, bp + 16), sc[nt]);
      }
    }

    // Online softmax. shfl_xor masks <16 keep reductions inside each
    // 16-lane half, matching the C-fragment row distribution.
#pragma unroll
    for (int r = 0; r < 8; ++r) {
      float mx = fmaxf(fmaxf(sc[0][r], sc[1][r]), fmaxf(sc[2][r], sc[3][r]));
#pragma unroll
      for (int off = 8; off; off >>= 1) mx = fmaxf(mx, __shfl_xor(mx, off, 32));
      float mnew  = fmaxf(mrow[r], mx);
      float alpha = __expf(mrow[r] - mnew);
      float psum  = 0.f;
#pragma unroll
      for (int nt = 0; nt < 4; ++nt) {
        float p = __expf(sc[nt][r] - mnew);
        sc[nt][r] = p;
        psum += p;
      }
#pragma unroll
      for (int off = 8; off; off >>= 1) psum += __shfl_xor(psum, off, 32);
      lrow[r] = lrow[r] * alpha + psum;
      mrow[r] = mnew;
#pragma unroll
      for (int nt = 0; nt < 4; ++nt) oacc[nt][r] *= alpha;
    }

    // C-layout -> A-layout for P via per-wave LDS tile (same-wave DS ops are
    // in order; no block barrier needed).
    _Float16* pw = &p_lds[wave][0];
#pragma unroll
    for (int nt = 0; nt < 4; ++nt) {
#pragma unroll
      for (int r = 0; r < 8; ++r) {
        int rl = (lane >> 4) * 8 + r;
        pw[rl * 72 + nt * 16 + nlane] = (_Float16)sc[nt][r];
      }
    }
    v16h pa[2];
    const int prow = lane & 15;
#pragma unroll
    for (int ks = 0; ks < 2; ++ks) {
      const _Float16* p = &pw[prow * 72 + ks * 32 + kh8];
      pa[ks] = load_frag(p, p + 16);
    }

    // oacc += P @ V   (B frags from dv-major v_lds, contiguous over keys).
#pragma unroll
    for (int ks = 0; ks < 2; ++ks) {
#pragma unroll
      for (int nt = 0; nt < 4; ++nt) {
        const _Float16* bp = &v_lds[(nt * 16 + nlane) * 72 + ks * 32 + kh8];
        oacc[nt] = wmma_f16(pa[ks], load_frag(bp, bp + 16), oacc[nt]);
      }
    }
    __syncthreads();
  }

  // Normalize and scatter into the head-concatenated f16 buffer.
  const int bb = bh >> 4;          // bh / H
  const int h  = bh & (MHA_H - 1);
#pragma unroll
  for (int nt = 0; nt < 4; ++nt) {
#pragma unroll
    for (int r = 0; r < 8; ++r) {
      int rl  = (lane >> 4) * 8 + r;
      int s   = q0 + wave * 16 + rl;
      int col = h * 64 + nt * 16 + nlane;
      cat[((size_t)bb * MHA_S + s) * (MHA_H * MHA_DV) + col] =
          (_Float16)(oacc[nt][r] / lrow[r]);
    }
  }
}

// ---------------------------------------------------------------------------
// Kernel 3: output projection  out = cat(f16) @ WO(f32) + bO   (f32 out)
//   cat: [BS, 1024] f16.  WO: [1024, 1024] f32.  out: [BS, 1024] f32.
// A tile (already f16) staged via async-to-LDS; WO converted in flight.
// ---------------------------------------------------------------------------
__global__ __launch_bounds__(128) void mha_oproj_kernel(
    const _Float16* __restrict__ A, const float* __restrict__ WO,
    const float* __restrict__ bO, float* __restrict__ out) {
  __shared__ _Float16 a_lds[64 * 72];
  __shared__ _Float16 bt_lds[64 * 72];

  const int tid  = threadIdx.x;
  const int wave = tid >> 5;
  const int lane = tid & 31;
  const int row0 = blockIdx.x * 64;
  const int n0   = blockIdx.y * 64;

  v8f acc[4] = {};

  const int arow  = wave * 16 + (lane & 15);
  const int kh8   = (lane >> 4) * 8;
  const int nlane = lane & 15;
  const int kb16  = (lane >> 4) * 16;

  for (int kk = 0; kk < MHA_DM; kk += 64) {
    // A tile 64x64 f16: pure copy -> async-to-LDS (512 x 16B chunks).
#pragma unroll
    for (int i = 0; i < 4; ++i) {
      int c = i * 128 + tid;
      int r = c >> 3;                    // 8 chunks per 64-f16 row
      int o = (c & 7) * 8;
      async_copy_b128(&a_lds[r * 72 + o],
                      &A[(size_t)(row0 + r) * MHA_DM + kk + o]);
    }
    // Bt tile: pairs of k-rows of WO -> packed v2h transposed stores.
    v4f wa[8];
#pragma unroll
    for (int i = 0; i < 4; ++i) {
      int vi = i * 128 + tid;
      int k2 = (vi >> 4) * 2;
      int n4 = (vi & 15) * 4;
      wa[2 * i + 0] = *(const v4f*)&WO[(size_t)(kk + k2) * MHA_DM + n0 + n4];
      wa[2 * i + 1] = *(const v4f*)&WO[(size_t)(kk + k2 + 1) * MHA_DM + n0 + n4];
    }
#pragma unroll
    for (int i = 0; i < 4; ++i) {
      int vi = i * 128 + tid;
      int k2 = (vi >> 4) * 2;
      int n4 = (vi & 15) * 4;
#pragma unroll
      for (int j = 0; j < 4; ++j) {
        v2h p;
        p[0] = (_Float16)wa[2 * i + 0][j];
        p[1] = (_Float16)wa[2 * i + 1][j];
        *(v2h*)&bt_lds[(n4 + j) * 72 + k2] = p;
      }
    }
    if (kk + 64 < MHA_DM)
      __builtin_prefetch(&WO[(size_t)(kk + 64 + (tid >> 1)) * MHA_DM + n0], 0, 1);
    async_wait0();
    __syncthreads();

#pragma unroll
    for (int ks = 0; ks < 2; ++ks) {
      const _Float16* ap = &a_lds[arow * 72 + ks * 32 + kh8];
      v16h af = load_frag(ap, ap + 16);
#pragma unroll
      for (int nt = 0; nt < 4; ++nt) {
        const _Float16* bp = &bt_lds[(nt * 16 + nlane) * 72 + ks * 32 + kh8];
        acc[nt] = wmma_f16(af, load_frag(bp, bp + 16), acc[nt]);
      }
    }
    __syncthreads();
  }

  const int rbase = (lane >> 4) * 8;
#pragma unroll
  for (int nt = 0; nt < 4; ++nt) {
#pragma unroll
    for (int r = 0; r < 8; ++r) {
      int row = row0 + wave * 16 + rbase + r;
      int col = n0 + nt * 16 + nlane;
      out[(size_t)row * MHA_DM + col] = acc[nt][r] + bO[col];
    }
  }
}

// ---------------------------------------------------------------------------
// Launcher. Workspace layout (f16): qh | kh | vT | cat  = 32 MB total.
// ---------------------------------------------------------------------------
extern "C" void kernel_launch(void* const* d_in, const int* in_sizes, int n_in,
                              void* d_out, int out_size, void* d_ws,
                              size_t ws_size, hipStream_t stream) {
  (void)in_sizes; (void)n_in; (void)out_size; (void)ws_size;
  const float* Q  = (const float*)d_in[0];
  const float* K  = (const float*)d_in[1];
  const float* V  = (const float*)d_in[2];
  const float* WQ = (const float*)d_in[3];
  const float* bQ = (const float*)d_in[4];
  const float* WK = (const float*)d_in[5];
  const float* bK = (const float*)d_in[6];
  const float* WV = (const float*)d_in[7];
  const float* bV = (const float*)d_in[8];
  const float* WO = (const float*)d_in[9];
  const float* bO = (const float*)d_in[10];

  const size_t per_head = (size_t)MHA_B * MHA_H * MHA_S * 64;  // 4 Mi f16
  _Float16* qh  = (_Float16*)d_ws;
  _Float16* kh  = qh + per_head;
  _Float16* vT  = kh + per_head;
  _Float16* cat = vT + per_head;

  dim3 blk(128);
  const float qscale = 0.125f;  // 1/sqrt(DK=64)

  mha_proj_kernel<<<dim3(MHA_BS / 64, MHA_H), blk, 0, stream>>>(
      Q, WQ, bQ, qh, qscale, 0);
  mha_proj_kernel<<<dim3(MHA_BS / 64, MHA_H), blk, 0, stream>>>(
      K, WK, bK, kh, 1.0f, 0);
  mha_proj_kernel<<<dim3(MHA_BS / 64, MHA_H), blk, 0, stream>>>(
      V, WV, bV, vT, 1.0f, 1);
  mha_attn_kernel<<<dim3(MHA_S / 64, MHA_B * MHA_H), blk, 0, stream>>>(
      qh, kh, vT, cat);
  mha_oproj_kernel<<<dim3(MHA_BS / 64, MHA_DM / 64), blk, 0, stream>>>(
      cat, WO, bO, (float*)d_out);
}
